// Int8OPTAttention_79259326481050
// MI455X (gfx1250) — 8-bit matrix operand order repaired and checked by reading, compiled, not run on hardware
//
#include <hip/hip_runtime.h>
#include <hip/hip_bf16.h>
#include <cstdint>


typedef __attribute__((ext_vector_type(8))) int  v8i;
typedef __attribute__((ext_vector_type(4))) unsigned int u32x4;
typedef __attribute__((ext_vector_type(8))) int  i32x8;
typedef __attribute__((ext_vector_type(4))) int  i32x4;

#define BSZ_   4
#define SEQ_   1024
#define EMB_   2048
#define HEADS_ 32
#define HDIM_  64

#define PROJ_A_ 4.0e-5f
#define PROJ_B_ 1.0f
#define QK_A_   (1.0f / (HDIM_ * 127.0f))
#define PV_A_   (1.0f / (127.0f * SEQ_))
#define OUT_A_  4.0e-5f
#define MASK_NEG_ (-1.0e9f)

union FragA { v8i v; unsigned long long d[4]; };
union FragB { v8i v; int4 q[2]; };

__device__ __forceinline__ v8i wmma_iu8(v8i a, v8i b, v8i c) {
    return __builtin_amdgcn_wmma_i32_16x16x64_iu8(true, a, true, b, c, false, false);
}

__device__ __forceinline__ signed char requant_i8(float y) {
    int q = __float2int_rn(y);
    q = q < -128 ? -128 : (q > 127 ? 127 : q);
    return (signed char)q;
}

#if defined(__gfx1250__) && __has_builtin(__builtin_amdgcn_tensor_load_to_lds)
#define HAS_TDM_ 1
__device__ __forceinline__ void tdm_load_2d(const void* gsrc, void* lds_dst,
                                            unsigned row_bytes,
                                            unsigned rows,
                                            unsigned tensor_w,
                                            unsigned tensor_h,
                                            unsigned stride_bytes)
{
    const unsigned long long ga = (unsigned long long)(uintptr_t)gsrc;
    u32x4 g0;
    g0.x = 1u;
    g0.y = (unsigned)(uintptr_t)lds_dst;
    g0.z = (unsigned)(ga & 0xffffffffu);
    g0.w = (unsigned)((ga >> 32) & 0x01ffffffu) | (2u << 30);
    i32x8 g1;
    g1[0] = 0;
    g1[1] = (int)((tensor_w & 0xffffu) << 16);
    g1[2] = (int)(((tensor_w >> 16) & 0xffffu) | ((tensor_h & 0xffffu) << 16));
    g1[3] = (int)(((tensor_h >> 16) & 0xffffu) | ((row_bytes & 0xffffu) << 16));
    g1[4] = (int)(rows & 0xffffu);
    g1[5] = (int)stride_bytes;
    g1[6] = 0;
    g1[7] = 0;
    const i32x4 gz = {0, 0, 0, 0};
#if __clang_major__ >= 23
    const i32x8 gz8 = {0, 0, 0, 0, 0, 0, 0, 0};
    __builtin_amdgcn_tensor_load_to_lds(g0, g1, gz, gz, gz8, 0);
#else
    __builtin_amdgcn_tensor_load_to_lds(g0, g1, gz, gz, 0);
#endif
}
#else
#define HAS_TDM_ 0
#endif

__global__ __launch_bounds__(256) void attn_i8_wmma(
    const signed char* __restrict__ Q,
    const signed char* __restrict__ Km,
    const signed char* __restrict__ Vm,
    signed char* __restrict__ O,
    float qk_a, float pv_a)
{
    extern __shared__ signed char smem[];
    signed char* Ks = smem;
    signed char* Vt = smem + 64 * 1024;
    signed char* Pb = smem + 128 * 1024;

    const int t    = threadIdx.x;
    const int lane = t & 31;
    const int wave = t >> 5;
    const int half = lane >> 4;
    const int l15  = lane & 15;

    const int rowblk = blockIdx.x & 7;
    const int bh     = blockIdx.x >> 3;
    const int h      = bh & (HEADS_ - 1);
    const int b      = bh >> 5;

    const size_t head_off = ((size_t)b * SEQ_) * EMB_ + (size_t)h * HDIM_;
    const signed char* Kp = Km + head_off;
    const signed char* Vp = Vm + head_off;

#if HAS_TDM_
    signed char* Vraw = smem + 160 * 1024;
    if (wave == 0) {
        tdm_load_2d(Kp, Ks,   HDIM_, SEQ_, EMB_, SEQ_, EMB_);
        tdm_load_2d(Vp, Vraw, HDIM_, SEQ_, EMB_, SEQ_, EMB_);
        __builtin_amdgcn_s_wait_tensorcnt(0);
    }
    __syncthreads();
    for (int it = 0; it < 16; ++it) {
        const int idx = it * 256 + t;
        const int row = idx >> 2;
        const int c16 = (idx & 3) * 16;
        int4 vv = *(const int4*)(Vraw + row * HDIM_ + c16);
        const signed char* vb = (const signed char*)&vv;
        #pragma unroll
        for (int j = 0; j < 16; ++j)
            Vt[(size_t)(c16 + j) * SEQ_ + row] = vb[j];
    }
#else
    for (int it = 0; it < 16; ++it) {
        const int idx = it * 256 + t;
        const int row = idx >> 2;
        const int c16 = (idx & 3) * 16;
        int4 kv = *(const int4*)(Kp + (size_t)row * EMB_ + c16);
        int4 vv = *(const int4*)(Vp + (size_t)row * EMB_ + c16);
        *(int4*)(Ks + row * HDIM_ + c16) = kv;
        const signed char* vb = (const signed char*)&vv;
        #pragma unroll
        for (int j = 0; j < 16; ++j)
            Vt[(size_t)(c16 + j) * SEQ_ + row] = vb[j];
    }
#endif
    __syncthreads();

    const int rowbase = rowblk * 128 + wave * 16;
    signed char* Pw = Pb + wave * (16 * 256);

    FragA qf;
    {
        const signed char* qrow = Q + head_off + (size_t)(rowbase + l15) * EMB_;
        #pragma unroll
        for (int c = 0; c < 4; ++c)
            qf.d[c] = *(const unsigned long long*)(qrow + (c >> 1) * 32 + half * 16 + (c & 1) * 8);
    }

    const v8i vzero = {0, 0, 0, 0, 0, 0, 0, 0};

    float mrun[8], lrun[8];
    #pragma unroll
    for (int r = 0; r < 8; ++r) { mrun[r] = -3.0e38f; lrun[r] = 0.0f; }

    const int st_end = rowblk * 8 + wave + 1;
    for (int st = 0; st < st_end; ++st) {
        FragB kf;
        const signed char* kb = Ks + (st * 16 + l15) * HDIM_;
        kf.q[0] = *(const int4*)(kb + half * 16);
        kf.q[1] = *(const int4*)(kb + 32 + half * 16);
        v8i c = wmma_iu8(qf.v, kf.v, vzero);
        const int scol = st * 16 + l15;
        #pragma unroll
        for (int r = 0; r < 8; ++r) {
            const int row = rowbase + r + half * 8;
            const float f = (scol <= row) ? qk_a * (float)c[r] : MASK_NEG_;
            const float mn = fmaxf(mrun[r], f);
            lrun[r] = lrun[r] * __expf(mrun[r] - mn) + __expf(f - mn);
            mrun[r] = mn;
        }
    }
    #pragma unroll
    for (int off = 1; off < 16; off <<= 1) {
        #pragma unroll
        for (int r = 0; r < 8; ++r) {
            const float mo = __shfl_xor(mrun[r], off, 32);
            const float lo = __shfl_xor(lrun[r], off, 32);
            const float mn = fmaxf(mrun[r], mo);
            lrun[r] = lrun[r] * __expf(mrun[r] - mn) + lo * __expf(mo - mn);
            mrun[r] = mn;
        }
    }
    float scl[8];
    #pragma unroll
    for (int r = 0; r < 8; ++r) scl[r] = 127.0f / lrun[r];

    v8i oacc[4];
    #pragma unroll
    for (int dt = 0; dt < 4; ++dt) oacc[dt] = vzero;

    const int sb_end = (rowbase + 15) / 256 + 1;
    for (int sb = 0; sb < sb_end; ++sb) {
        for (int stl = 0; stl < 16; ++stl) {
            const int st = sb * 16 + stl;
            FragB kf;
            const signed char* kb = Ks + (st * 16 + l15) * HDIM_;
            kf.q[0] = *(const int4*)(kb + half * 16);
            kf.q[1] = *(const int4*)(kb + 32 + half * 16);
            v8i c = wmma_iu8(qf.v, kf.v, vzero);
            const int scol = st * 16 + l15;
            #pragma unroll
            for (int r = 0; r < 8; ++r) {
                const int row = rowbase + r + half * 8;
                const float f = (scol <= row) ? qk_a * (float)c[r] : MASK_NEG_;
                const float p = __expf(f - mrun[r]) * scl[r];
                int pi = __float2int_rn(p);
                pi = pi < 0 ? 0 : (pi > 127 ? 127 : pi);
                Pw[(r + half * 8) * 256 + stl * 16 + l15] = (signed char)pi;
            }
        }
        for (int kcl = 0; kcl < 4; ++kcl) {
            FragA pf;
            const signed char* pbase = Pw + l15 * 256 + kcl * 64;
            #pragma unroll
            for (int c = 0; c < 4; ++c)
                pf.d[c] = *(const unsigned long long*)(pbase + (c >> 1) * 32 + half * 16 + (c & 1) * 8);
            const int kc = sb * 4 + kcl;
            #pragma unroll
            for (int dt = 0; dt < 4; ++dt) {
                FragB vf;
                const signed char* vbase = Vt + (size_t)(dt * 16 + l15) * SEQ_ + kc * 64;
                vf.q[0] = *(const int4*)(vbase + half * 16);
                vf.q[1] = *(const int4*)(vbase + 32 + half * 16);
                oacc[dt] = wmma_iu8(pf.v, vf.v, oacc[dt]);
            }
        }
    }

    #pragma unroll
    for (int dt = 0; dt < 4; ++dt) {
        const int d = dt * 16 + l15;
        #pragma unroll
        for (int r = 0; r < 8; ++r) {
            const int row = rowbase + r + half * 8;
            O[head_off + (size_t)row * EMB_ + d] = requant_i8(pv_a * (float)oacc[dt][r]);
        }
    }
}

__global__ __launch_bounds__(256) void gemm_i8_wmma(
    const signed char* __restrict__ X,
    const signed char* __restrict__ W,
    const void* __restrict__ bias,
    void* __restrict__ out,
    int M, int N, int K,
    float alpha, float beta, int epilogue)
{
    __shared__ signed char As[128 * 64];
    __shared__ signed char Bs[128 * 64];

    const int t    = threadIdx.x;
    const int lane = t & 31;
    const int wave = t >> 5;
    const int wm   = wave >> 2;
    const int wn   = wave & 3;
    const int half = lane >> 4;
    const int l15  = lane & 15;

    const int m0 = blockIdx.y * 128;
    const int n0 = blockIdx.x * 128;

    const v8i vzero = {0, 0, 0, 0, 0, 0, 0, 0};
    v8i acc[4][2];
    #pragma unroll
    for (int i = 0; i < 4; ++i)
        #pragma unroll
        for (int j = 0; j < 2; ++j) acc[i][j] = vzero;

    const int srow = t >> 1;
    const int soff = (t & 1) * 32;

    const signed char* xrow = X + (size_t)(m0 + srow) * K + soff;
    const signed char* wrow = W + (size_t)(n0 + srow) * K + soff;

    int4 ra0, ra1, rb0, rb1;
    auto load_panel = [&](int k0) {
        const int4* sa = (const int4*)(xrow + k0);
        ra0 = sa[0]; ra1 = sa[1];
        const int4* sb = (const int4*)(wrow + k0);
        rb0 = sb[0]; rb1 = sb[1];
    };
    load_panel(0);

    for (int k0 = 0; k0 < K; k0 += 64) {
        {
            int4* da = (int4*)(As + srow * 64 + soff);
            da[0] = ra0; da[1] = ra1;
            int4* db = (int4*)(Bs + srow * 64 + soff);
            db[0] = rb0; db[1] = rb1;
        }
        __syncthreads();

        if (k0 + 64 < K) {
            load_panel(k0 + 64);
            if (k0 + 128 < K) {
                __builtin_prefetch(xrow + k0 + 128, 0, 3);
                __builtin_prefetch(wrow + k0 + 128, 0, 3);
            }
        }

        FragA a[4];
        #pragma unroll
        for (int i = 0; i < 4; ++i) {
            const signed char* base = As + ((wm * 4 + i) * 16 + l15) * 64;
            #pragma unroll
            for (int c = 0; c < 4; ++c)
                a[i].d[c] = *(const unsigned long long*)(base + (c >> 1) * 32 + half * 16 + (c & 1) * 8);
        }
        FragB b[2];
        #pragma unroll
        for (int j = 0; j < 2; ++j) {
            const signed char* base = Bs + ((wn * 2 + j) * 16 + l15) * 64;
            b[j].q[0] = *(const int4*)(base + half * 16);
            b[j].q[1] = *(const int4*)(base + 32 + half * 16);
        }

        #pragma unroll
        for (int i = 0; i < 4; ++i)
            #pragma unroll
            for (int j = 0; j < 2; ++j)
                acc[i][j] = wmma_iu8(a[i].v, b[j].v, acc[i][j]);

        __syncthreads();
    }

    #pragma unroll
    for (int i = 0; i < 4; ++i) {
        #pragma unroll
        for (int j = 0; j < 2; ++j) {
            const int ng = n0 + (wn * 2 + j) * 16 + l15;
            #pragma unroll
            for (int r = 0; r < 8; ++r) {
                const int mg = m0 + (wm * 4 + i) * 16 + r + half * 8;
                float y = alpha * (float)acc[i][j][r];
                if (epilogue == 0) {
                    y += beta * (float)((const signed char*)bias)[ng];
                    ((signed char*)out)[(size_t)mg * N + ng] = requant_i8(y);
                } else {
                    ((float*)out)[(size_t)mg * N + ng] = y + ((const float*)bias)[ng];
                }
            }
        }
    }
}

extern "C" void kernel_launch(void* const* d_in, const int* in_sizes, int n_in,
                              void* d_out, int out_size, void* d_ws, size_t ws_size,
                              hipStream_t stream) {
    (void)in_sizes; (void)n_in; (void)out_size; (void)ws_size;

    const signed char* hs  = (const signed char*)d_in[0];
    const signed char* q_w = (const signed char*)d_in[1];
    const signed char* q_b = (const signed char*)d_in[2];
    const signed char* k_w = (const signed char*)d_in[3];
    const signed char* k_b = (const signed char*)d_in[4];
    const signed char* v_w = (const signed char*)d_in[5];
    const signed char* v_b = (const signed char*)d_in[6];
    const signed char* o_w = (const signed char*)d_in[7];
    const float*       o_b = (const float*)d_in[8];

    float* out = (float*)d_out;

    const size_t PROJ_BYTES = (size_t)BSZ_ * SEQ_ * EMB_;
    signed char* qbuf = (signed char*)d_ws;
    signed char* kbuf = qbuf + PROJ_BYTES;
    signed char* vbuf = kbuf + PROJ_BYTES;
    signed char* obuf = qbuf;

    const int M = BSZ_ * SEQ_;
    const int N = EMB_;
    const int K = EMB_;

    dim3 blk(256);
    dim3 gg(N / 128, M / 128);

    gemm_i8_wmma<<<gg, blk, 0, stream>>>(hs, q_w, q_b, qbuf, M, N, K, PROJ_A_, PROJ_B_, 0);
    gemm_i8_wmma<<<gg, blk, 0, stream>>>(hs, k_w, k_b, kbuf, M, N, K, PROJ_A_, PROJ_B_, 0);
    gemm_i8_wmma<<<gg, blk, 0, stream>>>(hs, v_w, v_b, vbuf, M, N, K, PROJ_A_, PROJ_B_, 0);

    const size_t smem = (size_t)224 * 1024;
    dim3 ga(BSZ_ * HEADS_ * (SEQ_ / 128));
    attn_i8_wmma<<<ga, blk, smem, stream>>>(qbuf, kbuf, vbuf, obuf, QK_A_, PV_A_);

    gemm_i8_wmma<<<gg, blk, 0, stream>>>(obuf, o_w, o_b, out, M, N, K, OUT_A_, 0.0f, 1);
}
